// MultiHeadSelfAttention_63617055588838
// MI455X (gfx1250) — matrix operand order repaired and checked by reading, not compiled, not run on hardware
//
#include <hip/hip_runtime.h>
#include <hip/hip_bf16.h>
#include <hip/hip_fp16.h>
#include <math.h>

typedef __attribute__((ext_vector_type(16))) _Float16 v16h;
typedef __attribute__((ext_vector_type(8)))  _Float16 v8h;
typedef __attribute__((ext_vector_type(8)))  float    v8f;

#define EMBED 512
#define NH    8
#define HD    64
#define BB    4
#define NN    2048
#define ATTN_SCALE 0.125f  // 64^-0.5

// ---------------------------------------------------------------------------
// WMMA helpers (CDNA5 f16 16x16x32, f32 accumulate)
// ---------------------------------------------------------------------------
static __device__ __forceinline__ v8f wmma_f16(v16h a, v16h b, v8f c) {
  return __builtin_amdgcn_wmma_f32_16x16x32_f16(
      /*neg_a=*/false, a, /*neg_b=*/false, b,
      /*c_mod=*/(short)0, c, /*reuse_a=*/false, /*reuse_b=*/false);
}

// A-matrix fragment: 16x32 f16 tile from row-major [M][K] storage.
// Lanes 0-15 -> M=lane, halves = K(k0+0..7), K(k0+16..23);
// lanes 16-31 -> M=lane-16, halves = K(k0+8..15), K(k0+24..31).
static __device__ __forceinline__ v16h load_a_frag(const _Float16* __restrict__ base,
                                                   int row0, int k0, int ld, int lane) {
  int m    = row0 + (lane & 15);
  int koff = (lane & 16) ? 8 : 0;
  const _Float16* p = base + (size_t)m * (size_t)ld + (size_t)(k0 + koff);
  union { v16h v; v8h h[2]; } u;
  u.h[0] = *(const v8h*)(p);
  u.h[1] = *(const v8h*)(p + 16);
  return u.v;
}

// Same A pattern from an LDS tile (stride in halves).
static __device__ __forceinline__ v16h load_a_frag_lds(const _Float16* p_base,
                                                       int ld, int lane) {
  int m    = lane & 15;
  int koff = (lane & 16) ? 8 : 0;
  const _Float16* p = p_base + m * ld + koff;
  union { v16h v; v8h h[2]; } u;
  u.h[0] = *(const v8h*)(p);
  u.h[1] = *(const v8h*)(p + 16);
  return u.v;
}

// B-matrix fragment: 32x16 f16 tile from B^T row-major [N][K] storage (K contig).
// Lanes 0-15 -> N=lane, K=k0+0..7,k0+16..23; lanes 16-31 -> N=lane-16, K=k0+8..15,k0+24..31.
static __device__ __forceinline__ v16h load_b_frag(const _Float16* __restrict__ base,
                                                   int col0, int k0, int ld, int lane) {
  int n    = col0 + (lane & 15);
  int koff = (lane & 16) ? 8 : 0;
  const _Float16* p = base + (size_t)n * (size_t)ld + (size_t)(k0 + koff);
  union { v16h v; v8h h[2]; } u;
  u.h[0] = *(const v8h*)(p);
  u.h[1] = *(const v8h*)(p + 16);
  return u.v;
}

// Same B pattern from an LDS tile (stride in halves).
static __device__ __forceinline__ v16h load_b_frag_lds(const _Float16* base,
                                                       int col0, int k0, int ld, int lane) {
  int n    = col0 + (lane & 15);
  int koff = (lane & 16) ? 8 : 0;
  const _Float16* p = base + n * ld + k0 + koff;
  union { v16h v; v8h h[2]; } u;
  u.h[0] = *(const v8h*)(p);
  u.h[1] = *(const v8h*)(p + 16);
  return u.v;
}

// ---------------------------------------------------------------------------
// CDNA5 async global->LDS copy (ASYNCcnt-tracked), GV addressing mode.
// VDST = VGPR holding LDS byte offset (low 32 bits of generic shared pointer).
// ---------------------------------------------------------------------------
static __device__ __forceinline__ void async_copy_b128(void* lds_dst, const void* gsrc) {
  unsigned lds_off = (unsigned)(uintptr_t)lds_dst;
  unsigned long long ga = (unsigned long long)(uintptr_t)gsrc;
  asm volatile("global_load_async_to_lds_b128 %0, %1, off"
               :: "v"(lds_off), "v"(ga) : "memory");
}
#define WAIT_ASYNCCNT_2() asm volatile("s_wait_asynccnt 0x2" ::: "memory")
#define WAIT_ASYNCCNT_0() asm volatile("s_wait_asynccnt 0x0" ::: "memory")

// ---------------------------------------------------------------------------
// Kernel 1: f32 -> f16 conversion
// ---------------------------------------------------------------------------
__global__ void __launch_bounds__(256)
cvt_f32_f16_kernel(const float* __restrict__ in, _Float16* __restrict__ out, int n) {
  int i = blockIdx.x * blockDim.x + threadIdx.x;
  if (i < n) out[i] = (_Float16)in[i];
}

// ---------------------------------------------------------------------------
// Kernel 2: QKV projection. Y[8192,1536] = Xh[8192,512] * Wh[1536,512]^T + b
// ---------------------------------------------------------------------------
__global__ void __launch_bounds__(256)
qkv_gemm_kernel(const _Float16* __restrict__ xh,
                const _Float16* __restrict__ wh,
                const float* __restrict__ bias,
                _Float16* __restrict__ qo,
                _Float16* __restrict__ ko,
                _Float16* __restrict__ vto) {
  const int lane = threadIdx.x & 31;
  const int wid  = threadIdx.x >> 5;
  const int tilesN = (3 * EMBED) / 32;           // 48
  int tile = blockIdx.x * 8 + wid;
  int tm = tile / tilesN, tn = tile % tilesN;
  int row0 = tm * 32, col0 = tn * 32;

  v8f acc[2][2];
  v8f zero = {};
  acc[0][0] = zero; acc[0][1] = zero; acc[1][0] = zero; acc[1][1] = zero;

  for (int k0 = 0; k0 < EMBED; k0 += 32) {
    v16h a0 = load_a_frag(xh, row0,      k0, EMBED, lane);
    v16h a1 = load_a_frag(xh, row0 + 16, k0, EMBED, lane);
    v16h b0 = load_b_frag(wh, col0,      k0, EMBED, lane);
    v16h b1 = load_b_frag(wh, col0 + 16, k0, EMBED, lane);
    acc[0][0] = wmma_f16(a0, b0, acc[0][0]);
    acc[0][1] = wmma_f16(a0, b1, acc[0][1]);
    acc[1][0] = wmma_f16(a1, b0, acc[1][0]);
    acc[1][1] = wmma_f16(a1, b1, acc[1][1]);
  }

  const int mhalf = (lane & 16) ? 8 : 0;
  const int nl = lane & 15;
#pragma unroll
  for (int i = 0; i < 2; ++i)
#pragma unroll
    for (int j = 0; j < 2; ++j) {
#pragma unroll
      for (int r = 0; r < 8; ++r) {
        int row = row0 + i * 16 + mhalf + r;
        int col = col0 + j * 16 + nl;
        float val = acc[i][j][r] + bias[col];
        int b = row >> 11, n = row & (NN - 1);
        int which = col >> 9;            // 0=Q 1=K 2=V
        int rem = col & (EMBED - 1);
        int h = rem >> 6, d = rem & (HD - 1);
        size_t bh = (size_t)(b * NH + h);
        _Float16 hv = (_Float16)val;
        if (which == 0)      qo[(bh * NN + n) * HD + d] = hv;
        else if (which == 1) ko[(bh * NN + n) * HD + d] = hv;
        else                 vto[(bh * HD + d) * NN + n] = hv;
      }
    }
}

// ---------------------------------------------------------------------------
// Kernel 3: flash attention with block-cooperative async K/V staging.
// 8 waves/block share one (b,h); each wave owns 16 query rows.
// Per 32-key block: async-copy K(32x64) + Vt(64x32) tiles to LDS (double
// buffered, ASYNCcnt-paced), S = Q K^T (4 wmma), masked online softmax,
// P staged through LDS, O += P V (4 wmma).
// ---------------------------------------------------------------------------
__global__ void __launch_bounds__(256)
attn_kernel(const _Float16* __restrict__ qh,
            const _Float16* __restrict__ kh,
            const _Float16* __restrict__ vth,
            const int* __restrict__ mask,
            _Float16* __restrict__ oh) {
  __shared__ __align__(16) _Float16 ldsK[2][32 * HD];   // 2 x 4KB: [key][d]
  __shared__ __align__(16) _Float16 ldsV[2][HD * 32];   // 2 x 4KB: [d][key]
  __shared__ __align__(16) _Float16 pbuf[8][16 * 32];   // 8KB P staging

  const int t    = threadIdx.x;
  const int lane = t & 31;
  const int wid  = t >> 5;
  int gw   = blockIdx.x * 8 + wid;
  int qblk = gw & ((NN / 16) - 1);   // 0..127
  int bh   = gw >> 7;                // constant across the block
  int b = bh >> 3;
  int h = bh & 7;
  int q0 = qblk * 16;

  const _Float16* qbase = qh  + (size_t)bh * NN * HD;
  const _Float16* kbase = kh  + (size_t)bh * NN * HD;
  const _Float16* vbase = vth + (size_t)bh * HD * NN;
  const int* mrow_base  = mask + ((size_t)b * NN + q0) * NN;

  // Q fragments stay resident (two k-steps cover D=64)
  v16h qa0 = load_a_frag(qbase, q0, 0,  HD, lane);
  v16h qa1 = load_a_frag(qbase, q0, 32, HD, lane);

  const int mhalf = (lane & 16) ? 8 : 0;
  const int nl = lane & 15;
  _Float16* myp = &pbuf[wid][0];

  float mrow[8], lrow[8];
  v8f zero = {};
  v8f oacc[4];
#pragma unroll
  for (int r = 0; r < 8; ++r) { mrow[r] = -1e30f; lrow[r] = 0.0f; }
#pragma unroll
  for (int d = 0; d < 4; ++d) oacc[d] = zero;

  // Per-thread stage addresses: K tile is 4KB contiguous in global;
  // Vt tile is 64 rows x 64B.
  const int vd = t >> 2, vpart = t & 3;

  // ---- prologue: stage j-block 0 into buffer 0 ----
  {
    const char* kg = (const char*)(kbase) + t * 16;
    async_copy_b128((char*)&ldsK[0][0] + t * 16, kg);
    const char* vg = (const char*)(vbase + (size_t)vd * NN) + vpart * 16;
    async_copy_b128((char*)&ldsV[0][0] + vd * 64 + vpart * 16, vg);
  }

  const int NBLK = NN / 32;   // 64 iterations, uniform across block
  for (int jb = 0; jb < NBLK; ++jb) {
    int j0 = jb * 32;
    int cur = jb & 1;
    // ---- stage next j-block into the other buffer ----
    if (jb + 1 < NBLK) {
      int nxt = (jb + 1) & 1;
      const char* kg = (const char*)(kbase + (size_t)(j0 + 32) * HD) + t * 16;
      async_copy_b128((char*)&ldsK[nxt][0] + t * 16, kg);
      const char* vg = (const char*)(vbase + (size_t)vd * NN + (j0 + 32)) + vpart * 16;
      async_copy_b128((char*)&ldsV[nxt][0] + vd * 64 + vpart * 16, vg);
      WAIT_ASYNCCNT_2();     // current stage (2 older ops) complete; next in flight
    } else {
      WAIT_ASYNCCNT_0();
    }
    __syncthreads();

    const _Float16* kt = &ldsK[cur][0];
    const _Float16* vt = &ldsV[cur][0];

    // ---- S = Q K^T for 16x32 key block (K fragments from LDS) ----
    v8f s0 = zero, s1 = zero;
    {
      v16h bk;
      bk = load_b_frag_lds(kt, 0,  0,  HD, lane); s0 = wmma_f16(qa0, bk, s0);
      bk = load_b_frag_lds(kt, 0,  32, HD, lane); s0 = wmma_f16(qa1, bk, s0);
      bk = load_b_frag_lds(kt, 16, 0,  HD, lane); s1 = wmma_f16(qa0, bk, s1);
      bk = load_b_frag_lds(kt, 16, 32, HD, lane); s1 = wmma_f16(qa1, bk, s1);
    }
    // ---- scale + mask ----
#pragma unroll
    for (int r = 0; r < 8; ++r) {
      int qr = mhalf + r;
      const int* mp = mrow_base + (size_t)qr * NN + j0;
      float v0 = s0[r] * ATTN_SCALE;
      float v1 = s1[r] * ATTN_SCALE;
      if (mp[nl] == 0)      v0 = -1e30f;
      if (mp[16 + nl] == 0) v1 = -1e30f;
      s0[r] = v0; s1[r] = v1;
    }
    // ---- online softmax (row stats via intra-16-lane xor shuffles) ----
#pragma unroll
    for (int r = 0; r < 8; ++r) {
      float mx = fmaxf(s0[r], s1[r]);
#pragma unroll
      for (int off = 1; off < 16; off <<= 1)
        mx = fmaxf(mx, __shfl_xor(mx, off, 32));
      float mnew  = fmaxf(mrow[r], mx);
      float alpha = __expf(mrow[r] - mnew);
      float p0 = __expf(s0[r] - mnew);
      float p1 = __expf(s1[r] - mnew);
      float rs = p0 + p1;
#pragma unroll
      for (int off = 1; off < 16; off <<= 1)
        rs += __shfl_xor(rs, off, 32);
      lrow[r] = lrow[r] * alpha + rs;
      mrow[r] = mnew;
#pragma unroll
      for (int d = 0; d < 4; ++d) oacc[d][r] *= alpha;
      int qr = mhalf + r;
      myp[qr * 32 + nl]      = (_Float16)p0;
      myp[qr * 32 + 16 + nl] = (_Float16)p1;
    }
    __builtin_amdgcn_wave_barrier();
    // ---- reload P in A-operand layout, O += P V (V fragments from LDS) ----
    v16h pa = load_a_frag_lds(myp, 32, lane);
    __builtin_amdgcn_wave_barrier();
#pragma unroll
    for (int d = 0; d < 4; ++d) {
      v16h bv = load_b_frag_lds(vt, d * 16, 0, 32, lane);
      oacc[d] = wmma_f16(pa, bv, oacc[d]);
    }
    __syncthreads();   // all waves done with this buffer before it is re-staged
  }

  // ---- normalize and store f16 O[B*N, 512] ----
#pragma unroll
  for (int r = 0; r < 8; ++r) {
    float inv = 1.0f / lrow[r];
    int qr = mhalf + r;
    size_t row = (size_t)b * NN + q0 + qr;
#pragma unroll
    for (int d = 0; d < 4; ++d) {
      float val = oacc[d][r] * inv;
      oh[row * EMBED + h * HD + d * 16 + nl] = (_Float16)val;
    }
  }
}

// ---------------------------------------------------------------------------
// Kernel 4: output projection. out[8192,512] = Oh * Wproj^T + b (f32 out)
// ---------------------------------------------------------------------------
__global__ void __launch_bounds__(256)
proj_gemm_kernel(const _Float16* __restrict__ ohm,
                 const _Float16* __restrict__ wh,
                 const float* __restrict__ bias,
                 float* __restrict__ out) {
  const int lane = threadIdx.x & 31;
  const int wid  = threadIdx.x >> 5;
  const int tilesN = EMBED / 32;                 // 16
  int tile = blockIdx.x * 8 + wid;
  int tm = tile / tilesN, tn = tile % tilesN;
  int row0 = tm * 32, col0 = tn * 32;

  v8f acc[2][2];
  v8f zero = {};
  acc[0][0] = zero; acc[0][1] = zero; acc[1][0] = zero; acc[1][1] = zero;

  for (int k0 = 0; k0 < EMBED; k0 += 32) {
    v16h a0 = load_a_frag(ohm, row0,      k0, EMBED, lane);
    v16h a1 = load_a_frag(ohm, row0 + 16, k0, EMBED, lane);
    v16h b0 = load_b_frag(wh,  col0,      k0, EMBED, lane);
    v16h b1 = load_b_frag(wh,  col0 + 16, k0, EMBED, lane);
    acc[0][0] = wmma_f16(a0, b0, acc[0][0]);
    acc[0][1] = wmma_f16(a0, b1, acc[0][1]);
    acc[1][0] = wmma_f16(a1, b0, acc[1][0]);
    acc[1][1] = wmma_f16(a1, b1, acc[1][1]);
  }

  const int mhalf = (lane & 16) ? 8 : 0;
  const int nl = lane & 15;
#pragma unroll
  for (int i = 0; i < 2; ++i)
#pragma unroll
    for (int j = 0; j < 2; ++j)
#pragma unroll
      for (int r = 0; r < 8; ++r) {
        int row = row0 + i * 16 + mhalf + r;
        int col = col0 + j * 16 + nl;
        out[(size_t)row * EMBED + col] = acc[i][j][r] + bias[col];
      }
}

// ---------------------------------------------------------------------------
// Host launcher
// ---------------------------------------------------------------------------
extern "C" void kernel_launch(void* const* d_in, const int* in_sizes, int n_in,
                              void* d_out, int out_size, void* d_ws, size_t ws_size,
                              hipStream_t stream) {
  const float* x      = (const float*)d_in[0];
  const float* qkv_w  = (const float*)d_in[1];
  const float* qkv_b  = (const float*)d_in[2];
  const float* proj_w = (const float*)d_in[3];
  const float* proj_b = (const float*)d_in[4];
  const int*   mask   = (const int*)d_in[5];
  float* out = (float*)d_out;

  char* ws = (char*)d_ws;
  size_t off = 0;
  _Float16* xh     = (_Float16*)(ws + off); off += (size_t)BB * NN * EMBED * 2;       // 8 MB
  _Float16* wqkvh  = (_Float16*)(ws + off); off += (size_t)3 * EMBED * EMBED * 2;     // 1.5 MB
  _Float16* wprojh = (_Float16*)(ws + off); off += (size_t)EMBED * EMBED * 2;         // 0.5 MB
  _Float16* qhp    = (_Float16*)(ws + off); off += (size_t)BB * NH * NN * HD * 2;     // 8 MB
  _Float16* khp    = (_Float16*)(ws + off); off += (size_t)BB * NH * NN * HD * 2;     // 8 MB
  _Float16* vthp   = (_Float16*)(ws + off); off += (size_t)BB * NH * HD * NN * 2;     // 8 MB
  _Float16* ohp    = (_Float16*)(ws + off); off += (size_t)BB * NN * EMBED * 2;       // 8 MB
  (void)ws_size; (void)off; (void)n_in; (void)in_sizes; (void)out_size;

  int n1 = BB * NN * EMBED;
  cvt_f32_f16_kernel<<<(n1 + 255) / 256, 256, 0, stream>>>(x, xh, n1);
  int n2 = 3 * EMBED * EMBED;
  cvt_f32_f16_kernel<<<(n2 + 255) / 256, 256, 0, stream>>>(qkv_w, wqkvh, n2);
  int n3 = EMBED * EMBED;
  cvt_f32_f16_kernel<<<(n3 + 255) / 256, 256, 0, stream>>>(proj_w, wprojh, n3);

  // QKV GEMM: 256 x 48 = 12288 tiles, 8 waves/block -> 1536 blocks
  qkv_gemm_kernel<<<1536, 256, 0, stream>>>(xh, wqkvh, qkv_b, qhp, khp, vthp);

  // Attention: 4*8*128 = 4096 waves -> 512 blocks
  attn_kernel<<<512, 256, 0, stream>>>(qhp, khp, vthp, mask, ohp);

  // Projection GEMM: 256 x 16 = 4096 tiles -> 512 blocks
  proj_gemm_kernel<<<512, 256, 0, stream>>>(ohp, wprojh, proj_b, out);
}
